// EPMoE_84782654423514
// MI455X (gfx1250) — matrix operand order repaired and checked by reading, not compiled, not run on hardware
//
#include <hip/hip_runtime.h>
#include <hip/hip_bf16.h>
#include <cmath>

// ---------------------------------------------------------------- constants
#define TT   4096
#define HH   1024
#define EE   8
#define DFF  4096
#define KSEL 2
#define NSLOTS (TT * KSEL)   // 8192 routed (token,expert) pairs, exactly

typedef __attribute__((ext_vector_type(16))) __bf16 bf16x16;
typedef __attribute__((ext_vector_type(8)))  __bf16 bf16x8;
typedef __attribute__((ext_vector_type(4)))  __bf16 bf16x4;
typedef __attribute__((ext_vector_type(8)))  float  f32x8;

#define CAT8_8(lo, hi) __builtin_shufflevector((lo), (hi), 0,1,2,3,4,5,6,7,8,9,10,11,12,13,14,15)

__device__ __forceinline__ int imin(int a, int b) { return a < b ? a : b; }

// ---------------------------------------------------------------- init
__global__ void moe_init_ws(int* counts, int* cursors) {
    if (threadIdx.x < EE) { counts[threadIdx.x] = 0; cursors[threadIdx.x] = 0; }
}

// ---------------------------------------------------------------- x -> bf16
__global__ __launch_bounds__(256) void moe_cvt_x(const float* __restrict__ x,
                                                 __bf16* __restrict__ xbf) {
    int i = (blockIdx.x * 256 + threadIdx.x) * 4;
    if (i >= TT * HH) return;
    float4 v = *(const float4*)(x + i);
    bf16x4 o;
    o[0] = (__bf16)v.x; o[1] = (__bf16)v.y; o[2] = (__bf16)v.z; o[3] = (__bf16)v.w;
    *(bf16x4*)(xbf + i) = o;
}

// ------------------------------------------- tiled transpose fp32 -> bf16
// in: [R][C] per expert, out: [C][R] per expert. R, C multiples of 32.
__global__ __launch_bounds__(256) void moe_transpose_cvt(const float* __restrict__ in,
                                                         __bf16* __restrict__ out,
                                                         int R, int C) {
    __shared__ float tile[32][33];
    const size_t per = (size_t)R * C;
    const float* src = in + per * blockIdx.z;
    __bf16*      dst = out + per * blockIdx.z;
    int tx = threadIdx.x & 31, ty = threadIdx.x >> 5;
    int c0 = blockIdx.x * 32, r0 = blockIdx.y * 32;
#pragma unroll
    for (int j = 0; j < 4; ++j)
        tile[ty + j * 8][tx] = src[(size_t)(r0 + ty + j * 8) * C + c0 + tx];
    __syncthreads();
#pragma unroll
    for (int j = 0; j < 4; ++j)
        dst[(size_t)(c0 + ty + j * 8) * R + r0 + tx] = (__bf16)tile[tx][ty + j * 8];
}

// ---------------------------------------------------------------- gate
__global__ __launch_bounds__(256) void moe_gate(const float* __restrict__ x,
                                                const float* __restrict__ gw,
                                                const float* __restrict__ gb,
                                                int* __restrict__ counts,
                                                int* __restrict__ tokE,
                                                float* __restrict__ tokW,
                                                float* __restrict__ partials) {
    const int tid = threadIdx.x;
    const int t = blockIdx.x * 256 + tid;

    float acc[EE];
#pragma unroll
    for (int e = 0; e < EE; ++e) acc[e] = gb[e];
    const float* xr = x + (size_t)t * HH;
    for (int h = 0; h < HH; h += 4) {
        float4 xv = *(const float4*)(xr + h);
        const float* gr = gw + (size_t)h * EE;
#pragma unroll
        for (int e = 0; e < EE; ++e)
            acc[e] += xv.x * gr[e] + xv.y * gr[EE + e] +
                      xv.z * gr[2 * EE + e] + xv.w * gr[3 * EE + e];
    }
    // softmax over E=8
    float m = acc[0];
#pragma unroll
    for (int e = 1; e < EE; ++e) m = fmaxf(m, acc[e]);
    float p[EE], s = 0.f;
#pragma unroll
    for (int e = 0; e < EE; ++e) { p[e] = expf(acc[e] - m); s += p[e]; }
    float inv = 1.f / s;
#pragma unroll
    for (int e = 0; e < EE; ++e) p[e] *= inv;
    // top-2, first index wins ties (jax top_k semantics)
    int e0 = 0;
#pragma unroll
    for (int e = 1; e < EE; ++e) if (p[e] > p[e0]) e0 = e;
    int e1 = (e0 == 0) ? 1 : 0;
#pragma unroll
    for (int e = 0; e < EE; ++e) if (e != e0 && p[e] > p[e1]) e1 = e;

    tokE[t * 2 + 0] = e0; tokE[t * 2 + 1] = e1;
    tokW[t * 2 + 0] = p[e0]; tokW[t * 2 + 1] = p[e1];
    atomicAdd(&counts[e0], 1);
    atomicAdd(&counts[e1], 1);

    // deterministic block reduction of 24 quantities:
    // q in [0,8): sum probs ; [8,16): sum mask ; [16,24): sum probs*mask
    __shared__ float red[256];
    float vals[24];
#pragma unroll
    for (int e = 0; e < EE; ++e) {
        float msk = (e == e0 || e == e1) ? 1.f : 0.f;
        vals[e] = p[e];
        vals[8 + e] = msk;
        vals[16 + e] = p[e] * msk;
    }
    for (int q = 0; q < 24; ++q) {
        red[tid] = vals[q];
        __syncthreads();
        for (int st = 128; st > 0; st >>= 1) {
            if (tid < st) red[tid] += red[tid + st];
            __syncthreads();
        }
        if (tid == 0) partials[blockIdx.x * 24 + q] = red[0];
        __syncthreads();
    }
}

// ------------------------------------- prefix offsets + aux loss (1 thread)
__global__ void moe_finalize_gate(const float* __restrict__ partials,
                                  const int* __restrict__ counts,
                                  int* __restrict__ offsets,
                                  float* __restrict__ aux_out) {
    if (threadIdx.x != 0 || blockIdx.x != 0) return;
    int run = 0;
    for (int e = 0; e < EE; ++e) { offsets[e] = run; run += counts[e]; }
    float imp[EE], us[EE], rw[EE];
    for (int e = 0; e < EE; ++e) {
        float a = 0.f, b = 0.f, c = 0.f;
        for (int blk = 0; blk < 16; ++blk) {
            a += partials[blk * 24 + e];
            b += partials[blk * 24 + 8 + e];
            c += partials[blk * 24 + 16 + e];
        }
        imp[e] = a; us[e] = b; rw[e] = c;
    }
    float mean = 0.f;
    for (int e = 0; e < EE; ++e) mean += imp[e];
    mean /= (float)EE;
    float var = 0.f;
    for (int e = 0; e < EE; ++e) { float d = imp[e] - mean; var += d * d; }
    var /= (float)(EE - 1);                      // unbiased, ddof=1
    float l1 = var / (float)(EE * EE);
    float l2 = 0.f;
    for (int e = 0; e < EE; ++e) l2 += (us[e] / (float)TT) * (rw[e] / (float)TT);
    l2 *= (float)EE;
    aux_out[0] = l1 + l2;
}

// ---------------------------------------------------------------- scatter
__global__ __launch_bounds__(256) void moe_scatter(const int* __restrict__ tokE,
                                                   const int* __restrict__ offsets,
                                                   int* __restrict__ cursors,
                                                   int* __restrict__ slotTok,
                                                   int* __restrict__ slotIdx) {
    int t = blockIdx.x * 256 + threadIdx.x;
#pragma unroll
    for (int k = 0; k < KSEL; ++k) {
        int e = tokE[t * 2 + k];
        int pos = atomicAdd(&cursors[e], 1);
        int slot = offsets[e] + pos;
        slotTok[slot] = t;
        slotIdx[t * 2 + k] = slot;
    }
}

// ---------------------------------------------------------------- WMMA GEMM
// Block tile 128(M) x 128(N); 8 waves as 4x2, wave tile 32x64 (8 WMMAs/k-step).
// A: bf16 rows of length KDIM (gathered token rows for stage 1, h rows stage 2).
// B: pre-transposed bf16, Bbase[e][n][k] — lane's 16 K-values are contiguous.
template <int KDIM, bool GATHER, bool RELU_BF16_OUT>
__global__ __launch_bounds__(256) void moe_gemm(const __bf16* __restrict__ Abase,
                                                const __bf16* __restrict__ Bbase,
                                                const float* __restrict__ bias,
                                                const int* __restrict__ counts,
                                                const int* __restrict__ offsets,
                                                const int* __restrict__ slotTok,
                                                __bf16* __restrict__ outBf,
                                                float* __restrict__ outF,
                                                int Ncols) {
    const int e = blockIdx.z;
    const int cnt = counts[e];
    const int mblock = blockIdx.y * 128;
    if (mblock >= cnt) return;
    const int base = offsets[e];
    const int nblock = blockIdx.x * 128;

    const int wave = threadIdx.x >> 5;
    const int lane = threadIdx.x & 31;
    const int lrow = lane & 15;      // row (A) / col (B,C) within 16
    const int khalf = lane >> 4;     // which K half-group this lane owns

    const int mw = mblock + (wave >> 1) * 32;   // expert-local row base of wave
    const int nw = nblock + (wave & 1) * 64;

    const __bf16* Bexp = Bbase + (size_t)e * Ncols * KDIM;

    const __bf16* Arow[2];
#pragma unroll
    for (int i = 0; i < 2; ++i) {
        int mloc = mw + i * 16 + lrow;
        int mc = imin(mloc, cnt - 1);
        size_t row = GATHER ? (size_t)slotTok[base + mc] : (size_t)(base + mc);
        Arow[i] = Abase + row * KDIM + khalf * 16;   // halves at +0 and +8
    }
    const __bf16* Brow[4];
#pragma unroll
    for (int j = 0; j < 4; ++j)
        Brow[j] = Bexp + (size_t)(nw + j * 16 + lrow) * KDIM + khalf * 16;

    f32x8 acc[2][4] = {};

    // k-loop; unroll 2 so the scheduler hoists step k+1 loads behind step k WMMAs
#pragma unroll 2
    for (int kt = 0; kt < KDIM; kt += 32) {
        bf16x16 a[2], b[4];
#pragma unroll
        for (int i = 0; i < 2; ++i) {
            bf16x8 lo = *(const bf16x8*)(Arow[i] + kt);
            bf16x8 hi = *(const bf16x8*)(Arow[i] + kt + 8);
            a[i] = CAT8_8(lo, hi);
            // pull the next A tile toward the caches (x/h rows are HBM-cold)
            if (kt + 64 < KDIM)
                __builtin_prefetch(Arow[i] + kt + 64, 0, 1);
        }
#pragma unroll
        for (int j = 0; j < 4; ++j)
            b[j] = *(const bf16x16*)(Brow[j] + kt);
#pragma unroll
        for (int i = 0; i < 2; ++i)
#pragma unroll
            for (int j = 0; j < 4; ++j)
                acc[i][j] = __builtin_amdgcn_wmma_f32_16x16x32_bf16(
                    false, a[i], false, b[j], (short)0, acc[i][j], false, false);
    }

    const float* biasE = bias + (size_t)e * Ncols;
    float bv[4];
#pragma unroll
    for (int j = 0; j < 4; ++j) bv[j] = biasE[nw + j * 16 + lrow];

    const bool fullTile = (mblock + 128 <= cnt);
    if (fullTile) {
        // fast path: every row in range, no per-element EXEC churn
#pragma unroll
        for (int i = 0; i < 2; ++i) {
#pragma unroll
            for (int j = 0; j < 4; ++j) {
                int nc = nw + j * 16 + lrow;
#pragma unroll
                for (int r = 0; r < 8; ++r) {
                    int mloc = mw + i * 16 + r + 8 * khalf;
                    float v = acc[i][j][r] + bv[j];
                    if (RELU_BF16_OUT)
                        outBf[(size_t)(base + mloc) * Ncols + nc] = (__bf16)fmaxf(v, 0.f);
                    else
                        outF[(size_t)(base + mloc) * Ncols + nc] = v;
                }
            }
        }
    } else {
#pragma unroll
        for (int i = 0; i < 2; ++i) {
#pragma unroll
            for (int j = 0; j < 4; ++j) {
                int nc = nw + j * 16 + lrow;
#pragma unroll
                for (int r = 0; r < 8; ++r) {
                    int mloc = mw + i * 16 + r + 8 * khalf;
                    if (mloc < cnt) {
                        float v = acc[i][j][r] + bv[j];
                        if (RELU_BF16_OUT)
                            outBf[(size_t)(base + mloc) * Ncols + nc] = (__bf16)fmaxf(v, 0.f);
                        else
                            outF[(size_t)(base + mloc) * Ncols + nc] = v;
                    }
                }
            }
        }
    }
}

// ---------------------------------------------------------------- combine
__global__ __launch_bounds__(256) void moe_combine(const float* __restrict__ ybuf,
                                                   const int* __restrict__ slotIdx,
                                                   const float* __restrict__ tokW,
                                                   float* __restrict__ out) {
    int idx = blockIdx.x * 256 + threadIdx.x;        // over T*H/4
    int t = idx / (HH / 4);
    int c = (idx % (HH / 4)) * 4;
    int s0 = slotIdx[t * 2 + 0], s1 = slotIdx[t * 2 + 1];
    float w0 = tokW[t * 2 + 0], w1 = tokW[t * 2 + 1];
    float4 y0 = *(const float4*)(ybuf + (size_t)s0 * HH + c);
    float4 y1 = *(const float4*)(ybuf + (size_t)s1 * HH + c);
    float4 o;
    o.x = w0 * y0.x + w1 * y1.x;
    o.y = w0 * y0.y + w1 * y1.y;
    o.z = w0 * y0.z + w1 * y1.z;
    o.w = w0 * y0.w + w1 * y1.w;
    *(float4*)(out + (size_t)t * HH + c) = o;
}

// ---------------------------------------------------------------- launch
extern "C" void kernel_launch(void* const* d_in, const int* in_sizes, int n_in,
                              void* d_out, int out_size, void* d_ws, size_t ws_size,
                              hipStream_t stream) {
    const float* x  = (const float*)d_in[0];
    const float* gw = (const float*)d_in[1];
    const float* gb = (const float*)d_in[2];
    const float* w1 = (const float*)d_in[3];
    const float* b1 = (const float*)d_in[4];
    const float* w2 = (const float*)d_in[5];
    const float* b2 = (const float*)d_in[6];
    float* out = (float*)d_out;

    char* ws = (char*)d_ws;
    size_t off = 0;
    auto take = [&](size_t bytes, size_t align) {
        off = (off + align - 1) & ~(align - 1);
        size_t o = off; off += bytes; return o;
    };
    size_t oCounts  = take(EE * 4, 64);
    size_t oCursors = take(EE * 4, 64);
    size_t oOffsets = take(EE * 4, 64);
    size_t oPart    = take(16 * 24 * 4, 128);
    size_t oTokE    = take((size_t)TT * 2 * 4, 128);
    size_t oTokW    = take((size_t)TT * 2 * 4, 128);
    size_t oSlotIdx = take((size_t)TT * 2 * 4, 128);
    size_t oSlotTok = take((size_t)NSLOTS * 4, 128);
    size_t oXbf     = take((size_t)TT * HH * 2, 128);
    size_t oW1t     = take((size_t)EE * HH * DFF * 2, 128);
    size_t oW2t     = take((size_t)EE * HH * DFF * 2, 128);
    size_t oHbuf    = take((size_t)NSLOTS * DFF * 2, 128);
    size_t oYbuf    = take((size_t)NSLOTS * HH * 4, 128);
    (void)ws_size; (void)in_sizes; (void)n_in; (void)out_size;

    int*    counts  = (int*)(ws + oCounts);
    int*    cursors = (int*)(ws + oCursors);
    int*    offsets = (int*)(ws + oOffsets);
    float*  part    = (float*)(ws + oPart);
    int*    tokE    = (int*)(ws + oTokE);
    float*  tokW    = (float*)(ws + oTokW);
    int*    slotIdx = (int*)(ws + oSlotIdx);
    int*    slotTok = (int*)(ws + oSlotTok);
    __bf16* xbf     = (__bf16*)(ws + oXbf);
    __bf16* w1t     = (__bf16*)(ws + oW1t);
    __bf16* w2t     = (__bf16*)(ws + oW2t);
    __bf16* hbuf    = (__bf16*)(ws + oHbuf);
    float*  ybuf    = (float*)(ws + oYbuf);

    moe_init_ws<<<1, 32, 0, stream>>>(counts, cursors);
    moe_cvt_x<<<(TT * HH / 4 + 255) / 256, 256, 0, stream>>>(x, xbf);
    // w1[e]: [H][DFF] -> w1t[e]: [DFF][H]
    moe_transpose_cvt<<<dim3(DFF / 32, HH / 32, EE), 256, 0, stream>>>(w1, w1t, HH, DFF);
    // w2[e]: [DFF][H] -> w2t[e]: [H][DFF]
    moe_transpose_cvt<<<dim3(HH / 32, DFF / 32, EE), 256, 0, stream>>>(w2, w2t, DFF, HH);

    moe_gate<<<TT / 256, 256, 0, stream>>>(x, gw, gb, counts, tokE, tokW, part);
    moe_finalize_gate<<<1, 32, 0, stream>>>(part, counts, offsets, out + (size_t)TT * HH);
    moe_scatter<<<TT / 256, 256, 0, stream>>>(tokE, offsets, cursors, slotTok, slotIdx);

    // Stage 1: h = relu(x @ w1[e] + b1[e]) over gathered tokens, bf16 out
    moe_gemm<HH, true, true><<<dim3(DFF / 128, TT / 128, EE), 256, 0, stream>>>(
        xbf, w1t, b1, counts, offsets, slotTok, hbuf, (float*)nullptr, DFF);
    // Stage 2: y = h @ w2[e] + b2[e], fp32 out
    moe_gemm<DFF, false, false><<<dim3(HH / 128, TT / 128, EE), 256, 0, stream>>>(
        hbuf, w2t, b2, counts, offsets, slotTok, (__bf16*)nullptr, ybuf, HH);

    moe_combine<<<TT * HH / 4 / 256, 256, 0, stream>>>(ybuf, slotIdx, tokW, out);
}
